// TwoTowerRecommender_22144851378180
// MI455X (gfx1250) — matrix operand order repaired and checked by reading, not compiled, not run on hardware
//
#include <hip/hip_runtime.h>
#include <hip/hip_bf16.h>

typedef __attribute__((ext_vector_type(16))) _Float16 v16h;
typedef __attribute__((ext_vector_type(8)))  _Float16 v8h;
typedef __attribute__((ext_vector_type(8)))  float    v8f;

#define N_USERS_C 100000
#define N_TIRES_C 100000
#define T_EVENTS_C 2000000

// ---------------------------------------------------------------------------
// A-fragment loaders for V_WMMA_F32_16X16X32_F16 (per CDNA5 ISA 16-bit A
// layout): lane<16 holds K{0..7,16..23}, lane>=16 holds K{8..15,24..31} of the
// current 32-wide K block. `Ar` already points at (row, kblock + base).
// ---------------------------------------------------------------------------
__device__ inline v16h loadA_f32(const float* __restrict__ Ar) {
  v8h c0, c1;
#pragma unroll
  for (int j = 0; j < 8; ++j) c0[j] = (_Float16)Ar[j];
#pragma unroll
  for (int j = 0; j < 8; ++j) c1[j] = (_Float16)Ar[16 + j];
  union { v16h v; v8h h[2]; } a;
  a.h[0] = c0; a.h[1] = c1;
  return a.v;
}

__device__ inline v16h loadA_f16(const _Float16* __restrict__ Ar) {
  union { v16h v; v8h h[2]; } a;
  a.h[0] = *(const v8h*)Ar;
  a.h[1] = *(const v8h*)(Ar + 16);
  return a.v;
}

__device__ inline v8f wmma_step(v16h a, const _Float16* __restrict__ Bfrag,
                                int kb, v8f acc) {
  v16h b = *(const v16h*)(Bfrag + (size_t)kb * 512);  // 32 lanes * 16 f16
  return __builtin_amdgcn_wmma_f32_16x16x32_f16(
      false, a, false, b, (short)0, acc, false, false);
}

// ---------------------------------------------------------------------------
// Pack a [K,N] f32 weight matrix into CDNA5 WMMA B-fragment order (f16).
// Packed element index = ((nb*KB + kb)*32 + lane)*16 + v.
// ---------------------------------------------------------------------------
__global__ void pack_w(const float* __restrict__ W, _Float16* __restrict__ out,
                       int K, int N) {
  int tid = blockIdx.x * 256 + threadIdx.x;
  if (tid >= K * N) return;
  int v    = tid & 15;
  int lane = (tid >> 4) & 31;
  int frag = tid >> 9;
  int KB = K >> 5;
  int kb = frag % KB;
  int nb = frag / KB;
  int n  = lane & 15;
  int kk = ((lane >> 4) << 3) + (v & 7) + ((v >> 3) << 4);
  out[tid] = (_Float16)W[((kb << 5) + kk) * N + (nb << 4) + n];
}

// start[u] = lower_bound(hist_segments, u); start[N_USERS] = E (sorted input)
__global__ void seg_starts(const int* __restrict__ seg, int* __restrict__ start,
                           int E, int U1) {
  int u = blockIdx.x * 256 + threadIdx.x;
  if (u >= U1) return;
  int lo = 0, hi = E;
  while (lo < hi) {
    int mid = (lo + hi) >> 1;
    if (seg[mid] < u) lo = mid + 1; else hi = mid;
  }
  start[u] = lo;
}

// History mean-pool, no atomics: 64 threads per user stream its event range
// with coalesced 256B reads of the L2-resident item_vec table. f16 out [U,64].
__global__ void hist_pool(const int* __restrict__ start,
                          const int* __restrict__ items,
                          const float* __restrict__ item_vec,
                          _Float16* __restrict__ pool) {
  int u   = blockIdx.x * 4 + (threadIdx.x >> 6);
  int col = threadIdx.x & 63;
  int s = start[u], e = start[u + 1];
  float acc = 0.f;
  for (int i = s; i < e; ++i)
    acc += item_vec[(size_t)items[i] * 64 + col];
  int cnt = e - s;
  float m = (cnt > 0) ? acc / (float)cnt : 0.f;
  pool[(size_t)u * 64 + col] = (_Float16)m;
}

// ---------------------------------------------------------------------------
// Generic WMMA GEMM: Y[M,N] = act(A[M,K] @ W[K,N] + bias)
//   NT = N/16; MODE 0: ReLU + f16 out; MODE 1: L2-normalize rows + f32 out
//   AF32: A is f32 (converted on load)
// ---------------------------------------------------------------------------
template <int NT, int MODE, bool AF32>
__global__ __launch_bounds__(256) void gemm_wmma(
    const void* __restrict__ Aq, int lda,
    const _Float16* __restrict__ Bp, const float* __restrict__ bias, int K,
    _Float16* __restrict__ outH, int ldh,
    float* __restrict__ outF) {
  constexpr int MT = 8 / NT;
  constexpr int BM = 16 * MT;
  const int lane = threadIdx.x & 31;
  const int wave = threadIdx.x >> 5;
  const int mt = wave / NT;
  const int nt = wave % NT;
  const int m0 = blockIdx.x * BM + mt * 16;
  const int KB = K >> 5;
  const int rowA = m0 + (lane & 15);
  const int base = (lane < 16) ? 0 : 8;
  const int col = lane & 15;
  const int halfsel = lane >> 4;

  v8f acc;
  {
    float bv = bias[nt * 16 + col];
#pragma unroll
    for (int r = 0; r < 8; ++r) acc[r] = bv;
  }

  const _Float16* Bfrag = Bp + ((size_t)nt * KB * 32 + lane) * 16;

  for (int kb = 0; kb < KB; ++kb) {
    v16h a;
    if (AF32)
      a = loadA_f32((const float*)Aq + (size_t)rowA * lda + kb * 32 + base);
    else
      a = loadA_f16((const _Float16*)Aq + (size_t)rowA * lda + kb * 32 + base);
    acc = wmma_step(a, Bfrag, kb, acc);
  }

  if constexpr (MODE == 0) {
#pragma unroll
    for (int r = 0; r < 8; ++r) {
      float v = acc[r];
      v = v > 0.f ? v : 0.f;
      int row = m0 + r + halfsel * 8;
      outH[(size_t)row * ldh + nt * 16 + col] = (_Float16)v;
    }
  } else {
    __shared__ float tile[32 * 65];
    __shared__ float inv[32];
#pragma unroll
    for (int r = 0; r < 8; ++r)
      tile[(mt * 16 + r + halfsel * 8) * 65 + nt * 16 + col] = acc[r];
    __syncthreads();
    if (threadIdx.x < 32) {
      int rr = threadIdx.x;
      float s = 0.f;
#pragma unroll
      for (int i = 0; i < 64; ++i) { float x = tile[rr * 65 + i]; s += x * x; }
      inv[rr] = 1.0f / fmaxf(sqrtf(s), 1e-12f);
    }
    __syncthreads();
    const int rowsBase = blockIdx.x * 32;
#pragma unroll
    for (int j = 0; j < 8; ++j) {
      int e = threadIdx.x + j * 256;
      int rr = e >> 6, cc = e & 63;
      outF[(size_t)(rowsBase + rr) * 64 + cc] = tile[rr * 65 + cc] * inv[rr];
    }
  }
}

// ---------------------------------------------------------------------------
// Fused item-tower layer 1: h1 = relu(concat(h_tire, h_brand[bidx],
// h_size[sidx], specs, text_proj) @ W_i1 + b_i1), K=512, N=128.
// The concat is never materialized: every 8-wide A chunk lies in one source
// (section boundaries 128/256/384/448 are 32-aligned) and the section choice
// depends only on kb, so the K loop splits into 5 uniform sub-loops.
// ---------------------------------------------------------------------------
__global__ __launch_bounds__(256) void gemm_item1(
    const float* __restrict__ h_tire, const float* __restrict__ h_brand,
    const float* __restrict__ h_size, const float* __restrict__ specs,
    const _Float16* __restrict__ textp,
    const int* __restrict__ bidx, const int* __restrict__ sidx,
    const _Float16* __restrict__ Bp, const float* __restrict__ bias,
    _Float16* __restrict__ outH) {
  const int lane = threadIdx.x & 31;
  const int nt = threadIdx.x >> 5;        // 8 waves cover N=128
  const int m0 = blockIdx.x * 16;
  const int rowA = m0 + (lane & 15);
  const int base = (lane < 16) ? 0 : 8;
  const int col = lane & 15;
  const int halfsel = lane >> 4;
  constexpr int KB = 16;

  const int bi = bidx[rowA];
  const int si = sidx[rowA];

  v8f acc;
  {
    float bv = bias[nt * 16 + col];
#pragma unroll
    for (int r = 0; r < 8; ++r) acc[r] = bv;
  }

  const _Float16* Bfrag = Bp + ((size_t)nt * KB * 32 + lane) * 16;

  const float* At = h_tire + (size_t)rowA * 128 + base;      // K 0..127
#pragma unroll
  for (int s = 0; s < 4; ++s)
    acc = wmma_step(loadA_f32(At + s * 32), Bfrag, s, acc);

  const float* Ab = h_brand + (size_t)bi * 128 + base;       // K 128..255
#pragma unroll
  for (int s = 0; s < 4; ++s)
    acc = wmma_step(loadA_f32(Ab + s * 32), Bfrag, 4 + s, acc);

  const float* As = h_size + (size_t)si * 128 + base;        // K 256..383
#pragma unroll
  for (int s = 0; s < 4; ++s)
    acc = wmma_step(loadA_f32(As + s * 32), Bfrag, 8 + s, acc);

  const float* Ap = specs + (size_t)rowA * 64 + base;        // K 384..447
#pragma unroll
  for (int s = 0; s < 2; ++s)
    acc = wmma_step(loadA_f32(Ap + s * 32), Bfrag, 12 + s, acc);

  const _Float16* Ax = textp + (size_t)rowA * 64 + base;     // K 448..511
#pragma unroll
  for (int s = 0; s < 2; ++s)
    acc = wmma_step(loadA_f16(Ax + s * 32), Bfrag, 14 + s, acc);

#pragma unroll
  for (int r = 0; r < 8; ++r) {
    float v = acc[r];
    v = v > 0.f ? v : 0.f;
    int row = m0 + r + halfsel * 8;
    outH[(size_t)row * 128 + nt * 16 + col] = (_Float16)v;
  }
}

// ---------------------------------------------------------------------------
// Fused user-tower layer 1: h1 = relu(concat(h_user, history_pool) @ W_u1
// + b_u1), K=192, N=128. Sections: K 0..127 f32, K 128..191 f16.
// ---------------------------------------------------------------------------
__global__ __launch_bounds__(256) void gemm_user1(
    const float* __restrict__ h_user, const _Float16* __restrict__ pool,
    const _Float16* __restrict__ Bp, const float* __restrict__ bias,
    _Float16* __restrict__ outH) {
  const int lane = threadIdx.x & 31;
  const int nt = threadIdx.x >> 5;
  const int m0 = blockIdx.x * 16;
  const int rowA = m0 + (lane & 15);
  const int base = (lane < 16) ? 0 : 8;
  const int col = lane & 15;
  const int halfsel = lane >> 4;
  constexpr int KB = 6;

  v8f acc;
  {
    float bv = bias[nt * 16 + col];
#pragma unroll
    for (int r = 0; r < 8; ++r) acc[r] = bv;
  }

  const _Float16* Bfrag = Bp + ((size_t)nt * KB * 32 + lane) * 16;

  const float* Au = h_user + (size_t)rowA * 128 + base;      // K 0..127
#pragma unroll
  for (int s = 0; s < 4; ++s)
    acc = wmma_step(loadA_f32(Au + s * 32), Bfrag, s, acc);

  const _Float16* Apl = pool + (size_t)rowA * 64 + base;     // K 128..191
#pragma unroll
  for (int s = 0; s < 2; ++s)
    acc = wmma_step(loadA_f16(Apl + s * 32), Bfrag, 4 + s, acc);

#pragma unroll
  for (int r = 0; r < 8; ++r) {
    float v = acc[r];
    v = v > 0.f ? v : 0.f;
    int row = m0 + r + halfsel * 8;
    outH[(size_t)row * 128 + nt * 16 + col] = (_Float16)v;
  }
}

// ---------------------------------------------------------------------------
extern "C" void kernel_launch(void* const* d_in, const int* in_sizes, int n_in,
                              void* d_out, int out_size, void* d_ws, size_t ws_size,
                              hipStream_t stream) {
  const float* h_user = (const float*)d_in[0];
  const float* h_tire = (const float*)d_in[1];
  const float* h_brand = (const float*)d_in[2];
  const float* h_size = (const float*)d_in[3];
  const float* specs = (const float*)d_in[4];
  const float* text = (const float*)d_in[5];
  const int* bidx = (const int*)d_in[6];
  const int* sidx = (const int*)d_in[7];
  const int* hitems = (const int*)d_in[8];
  const int* hseg = (const int*)d_in[9];
  const float* W_tp = (const float*)d_in[10]; const float* b_tp = (const float*)d_in[11];
  const float* W_i1 = (const float*)d_in[12]; const float* b_i1 = (const float*)d_in[13];
  const float* W_i2 = (const float*)d_in[14]; const float* b_i2 = (const float*)d_in[15];
  const float* W_i3 = (const float*)d_in[16]; const float* b_i3 = (const float*)d_in[17];
  const float* W_u1 = (const float*)d_in[18]; const float* b_u1 = (const float*)d_in[19];
  const float* W_u2 = (const float*)d_in[20]; const float* b_u2 = (const float*)d_in[21];
  const float* W_u3 = (const float*)d_in[22]; const float* b_u3 = (const float*)d_in[23];

  char* ws = (char*)d_ws;
  size_t off = 0;
  auto alloc = [&](size_t bytes) -> void* {
    void* p = ws + off;
    off += bytes;
    off = (off + 255) & ~(size_t)255;
    return p;
  };

  _Float16* Bp_tp = (_Float16*)alloc(384 * 64 * 2);
  _Float16* Bp_i1 = (_Float16*)alloc(512 * 128 * 2);
  _Float16* Bp_i2 = (_Float16*)alloc(128 * 128 * 2);
  _Float16* Bp_i3 = (_Float16*)alloc(128 * 64 * 2);
  _Float16* Bp_u1 = (_Float16*)alloc(192 * 128 * 2);
  _Float16* Bp_u2 = (_Float16*)alloc(128 * 128 * 2);
  _Float16* Bp_u3 = (_Float16*)alloc(128 * 64 * 2);
  _Float16* textp = (_Float16*)alloc((size_t)N_TIRES_C * 64 * 2);
  _Float16* pool  = (_Float16*)alloc((size_t)N_USERS_C * 64 * 2);
  _Float16* h1    = (_Float16*)alloc((size_t)N_TIRES_C * 128 * 2);
  _Float16* h2    = (_Float16*)alloc((size_t)N_TIRES_C * 128 * 2);
  int* start = (int*)alloc((size_t)(N_USERS_C + 1) * 4);

  float* user_vec = (float*)d_out;
  float* item_vec = (float*)d_out + (size_t)N_USERS_C * 64;

  // --- weight repack (f32 -> f16 WMMA B-fragments) ---
  pack_w<<<(384 * 64) / 256, 256, 0, stream>>>(W_tp, Bp_tp, 384, 64);
  pack_w<<<(512 * 128) / 256, 256, 0, stream>>>(W_i1, Bp_i1, 512, 128);
  pack_w<<<(128 * 128) / 256, 256, 0, stream>>>(W_i2, Bp_i2, 128, 128);
  pack_w<<<(128 * 64) / 256, 256, 0, stream>>>(W_i3, Bp_i3, 128, 64);
  pack_w<<<(192 * 128) / 256, 256, 0, stream>>>(W_u1, Bp_u1, 192, 128);
  pack_w<<<(128 * 128) / 256, 256, 0, stream>>>(W_u2, Bp_u2, 128, 128);
  pack_w<<<(128 * 64) / 256, 256, 0, stream>>>(W_u3, Bp_u3, 128, 64);

  seg_starts<<<(N_USERS_C + 1 + 255) / 256, 256, 0, stream>>>(
      hseg, start, T_EVENTS_C, N_USERS_C + 1);

  // --- item tower ---
  gemm_wmma<4, 0, true><<<N_TIRES_C / 32, 256, 0, stream>>>(
      text, 384, Bp_tp, b_tp, 384, textp, 64, nullptr);           // text proj
  gemm_item1<<<N_TIRES_C / 16, 256, 0, stream>>>(
      h_tire, h_brand, h_size, specs, textp, bidx, sidx, Bp_i1, b_i1, h1);
  gemm_wmma<8, 0, false><<<N_TIRES_C / 16, 256, 0, stream>>>(
      h1, 128, Bp_i2, b_i2, 128, h2, 128, nullptr);
  gemm_wmma<4, 1, false><<<N_TIRES_C / 32, 256, 0, stream>>>(
      h2, 128, Bp_i3, b_i3, 128, nullptr, 0, item_vec);

  // --- history mean-pool (item_vec stays L2-resident) ---
  hist_pool<<<N_USERS_C / 4, 256, 0, stream>>>(start, hitems, item_vec, pool);

  // --- user tower ---
  gemm_user1<<<N_USERS_C / 16, 256, 0, stream>>>(
      h_user, pool, Bp_u1, b_u1, h1);
  gemm_wmma<8, 0, false><<<N_USERS_C / 16, 256, 0, stream>>>(
      h1, 128, Bp_u2, b_u2, 128, h2, 128, nullptr);
  gemm_wmma<4, 1, false><<<N_USERS_C / 32, 256, 0, stream>>>(
      h2, 128, Bp_u3, b_u3, 128, nullptr, 0, user_vec);
}
